// DualStreamTemporalModel_86715389706356
// MI455X (gfx1250) — compile-verified, matrix operand order checked by reading, not run on hardware
//
#include <hip/hip_runtime.h>
#include <hip/hip_bf16.h>
#include <math.h>

typedef __attribute__((ext_vector_type(16))) __bf16 v16bf;
typedef __attribute__((ext_vector_type(8)))  __bf16 v8bf;
typedef __attribute__((ext_vector_type(8)))  float  v8f;

#define BB   4
#define TT   2048
#define INF_ 64
#define HH   256
#define DD   512
#define GATES 1024

__device__ __forceinline__ float sigm_(float x) { return 1.0f / (1.0f + expf(-x)); }

// WMMA 16-bit A/B fragment (ISA 7.12.2): lane covers one row; elements
// [0..7] = K k0..k0+7, [8..15] = K k0+16..k0+23, k0 = kb*32 + (lane>=16)*8.
__device__ __forceinline__ v16bf frag_bf16(const __bf16* __restrict__ p, int k0) {
  v8bf a = *(const v8bf*)(p + k0);
  v8bf b = *(const v8bf*)(p + k0 + 16);
  v16bf r;
#pragma unroll
  for (int j = 0; j < 8; ++j) { r[j] = a[j]; r[j + 8] = b[j]; }
  return r;
}

// Same, pointer already offset to k0 (for pointer-increment pipelines).
__device__ __forceinline__ v16bf frag_bf16_p(const __bf16* __restrict__ p) {
  v8bf a = *(const v8bf*)(p);
  v8bf b = *(const v8bf*)(p + 16);
  v16bf r;
#pragma unroll
  for (int j = 0; j < 8; ++j) { r[j] = a[j]; r[j + 8] = b[j]; }
  return r;
}

// Async global -> LDS copy (ASYNCcnt-tracked), CDNA5 GLOBAL_LOAD_ASYNC_TO_LDS_B128.
__device__ __forceinline__ void async_copy_b128(unsigned lds_addr, const float* gptr) {
  asm volatile("global_load_async_to_lds_b128 %0, %1, off"
               :: "v"(lds_addr), "v"(gptr) : "memory");
}

// ---------------------------------------------------------------------------
// Packing kernels: one-time f32 -> bf16 conversions (activations & weights).
// ---------------------------------------------------------------------------
__global__ void pack_bf16(const float* __restrict__ src, __bf16* __restrict__ dst, int n) {
  int i = blockIdx.x * blockDim.x + threadIdx.x;
  if (i < n) dst[i] = (__bf16)src[i];
}

// conv_w (H, IN, K=5) -> bf16 [kk][h][i] (contiguous over i for WMMA B-frags)
__global__ void pack_convw(const float* __restrict__ src, __bf16* __restrict__ dst) {
  int i = blockIdx.x * blockDim.x + threadIdx.x;
  if (i < 5 * HH * INF_) {
    int kk = i / (HH * INF_);
    int r  = i % (HH * INF_);
    int h  = r / INF_, c = r % INF_;
    dst[i] = (__bf16)src[h * (INF_ * 5) + c * 5 + kk];
  }
}

// ---------------------------------------------------------------------------
// C[M,N] = A[M,K] @ W[N,K]^T + bias1 + bias2, all-bf16, f32 acc.
// One wave: one M-tile x 4 consecutive N-tiles (A reused in regs).
// Ping-pong software pipeline (K/32 must be even): set-1 loads overlap set-0
// wmmas and vice versa, with no register-rotation moves.
// ---------------------------------------------------------------------------
__global__ void gemm_bf16_wmma(const __bf16* __restrict__ A, int lda,
                               const __bf16* __restrict__ W, int ldb,
                               const float* __restrict__ bias1,
                               const float* __restrict__ bias2,
                               float* __restrict__ C, int ldc,
                               int Mtiles, int Ntiles, int K) {
  int wave = blockIdx.x * (blockDim.x >> 5) + (threadIdx.x >> 5);
  int ngrp = Ntiles >> 2;
  if (wave >= Mtiles * ngrp) return;
  int ng = wave % ngrp, mt = wave / ngrp;
  int lane = threadIdx.x & 31, lo = lane & 15, hi = lane >> 4;

  const __bf16* pa  = A + (size_t)(mt * 16 + lo) * lda + hi * 8;
  const __bf16* pb0 = W + (size_t)((ng * 4 + 0) * 16 + lo) * ldb + hi * 8;
  const __bf16* pb1 = W + (size_t)((ng * 4 + 1) * 16 + lo) * ldb + hi * 8;
  const __bf16* pb2 = W + (size_t)((ng * 4 + 2) * 16 + lo) * ldb + hi * 8;
  const __bf16* pb3 = W + (size_t)((ng * 4 + 3) * 16 + lo) * ldb + hi * 8;

  int KB = K >> 5;                       // even for all call sites (2, 8, 16)
  v8f acc[4] = {};

  v16bf a0  = frag_bf16_p(pa);
  v16bf c00 = frag_bf16_p(pb0);
  v16bf c01 = frag_bf16_p(pb1);
  v16bf c02 = frag_bf16_p(pb2);
  v16bf c03 = frag_bf16_p(pb3);

  for (int kb = 0; kb < KB; kb += 2) {
    // stage B loads (kb+1) overlap stage A wmmas (kb)
    v16bf a1  = frag_bf16_p(pa + 32);
    v16bf c10 = frag_bf16_p(pb0 + 32);
    v16bf c11 = frag_bf16_p(pb1 + 32);
    v16bf c12 = frag_bf16_p(pb2 + 32);
    v16bf c13 = frag_bf16_p(pb3 + 32);
    __builtin_prefetch(pa + 64, 0, 3);
    __builtin_prefetch(pb0 + 64, 0, 3);

    acc[0] = __builtin_amdgcn_wmma_f32_16x16x32_bf16(false, a0, false, c00, (short)0, acc[0], false, false);
    acc[1] = __builtin_amdgcn_wmma_f32_16x16x32_bf16(false, a0, false, c01, (short)0, acc[1], false, false);
    acc[2] = __builtin_amdgcn_wmma_f32_16x16x32_bf16(false, a0, false, c02, (short)0, acc[2], false, false);
    acc[3] = __builtin_amdgcn_wmma_f32_16x16x32_bf16(false, a0, false, c03, (short)0, acc[3], false, false);

    pa += 64; pb0 += 64; pb1 += 64; pb2 += 64; pb3 += 64;
    if (kb + 2 < KB) {                   // stage A loads (kb+2) overlap stage B wmmas
      a0  = frag_bf16_p(pa);
      c00 = frag_bf16_p(pb0);
      c01 = frag_bf16_p(pb1);
      c02 = frag_bf16_p(pb2);
      c03 = frag_bf16_p(pb3);
    }

    acc[0] = __builtin_amdgcn_wmma_f32_16x16x32_bf16(false, a1, false, c10, (short)0, acc[0], false, false);
    acc[1] = __builtin_amdgcn_wmma_f32_16x16x32_bf16(false, a1, false, c11, (short)0, acc[1], false, false);
    acc[2] = __builtin_amdgcn_wmma_f32_16x16x32_bf16(false, a1, false, c12, (short)0, acc[2], false, false);
    acc[3] = __builtin_amdgcn_wmma_f32_16x16x32_bf16(false, a1, false, c13, (short)0, acc[3], false, false);
  }

#pragma unroll
  for (int u = 0; u < 4; ++u) {
    int col = (ng * 4 + u) * 16 + lo;
    float badd = (bias1 ? bias1[col] : 0.f) + (bias2 ? bias2[col] : 0.f);
#pragma unroll
    for (int r = 0; r < 8; ++r) {
      int row = mt * 16 + r + 8 * hi;
      C[(size_t)row * ldc + col] = acc[u][r] + badd;
    }
  }
}

// ---------------------------------------------------------------------------
// Conv1d(K=5, pad=2) as 5 shifted GEMMs + BN + SiLU.
// Writes merged[:,256:512] (f32) and merged_bf[:,256:512] (bf16).
// ---------------------------------------------------------------------------
__global__ void conv_bn_silu_wmma(const __bf16* __restrict__ xbf,
                                  const __bf16* __restrict__ convw_bf,  // [kk][ch][i]
                                  const float* __restrict__ conv_b,
                                  const float* __restrict__ bn_g,
                                  const float* __restrict__ bn_b,
                                  const float* __restrict__ bn_mean,
                                  const float* __restrict__ bn_var,
                                  float* __restrict__ merged,
                                  __bf16* __restrict__ merged_bf) {
  int wave = blockIdx.x * (blockDim.x >> 5) + (threadIdx.x >> 5);
  int nt = wave & 15, mt = wave >> 4;
  int lane = threadIdx.x & 31, lo = lane & 15, hi = lane >> 4;

  int btA = mt * 16 + lo;              // A row (b*T+t); tiles never cross batches
  int tA  = btA & (TT - 1);
  int ch  = nt * 16 + lo;              // output channel (B row & store col)

  v8f acc = {};
#pragma unroll
  for (int kk = 0; kk < 5; ++kk) {
    int sh = kk - 2;
    bool valid = ((unsigned)(tA + sh) < (unsigned)TT);
    const __bf16* arow = xbf + (size_t)(valid ? (btA + sh) : btA) * INF_;
    const __bf16* brow = convw_bf + (size_t)kk * (HH * INF_) + (size_t)ch * INF_;
#pragma unroll
    for (int kb = 0; kb < 2; ++kb) {
      int k0 = kb * 32 + hi * 8;
      v16bf a = frag_bf16(arow, k0);
      if (!valid) {
#pragma unroll
        for (int j = 0; j < 16; ++j) a[j] = (__bf16)0.0f;
      }
      v16bf b = frag_bf16(brow, k0);
      acc = __builtin_amdgcn_wmma_f32_16x16x32_bf16(false, a, false, b, (short)0, acc, false, false);
    }
  }

  float scale = bn_g[ch] * rsqrtf(bn_var[ch] + 1e-5f);
  float shift = bn_b[ch] - bn_mean[ch] * scale;
  float cb = conv_b[ch];
#pragma unroll
  for (int r = 0; r < 8; ++r) {
    int row = mt * 16 + r + 8 * hi;
    float y = (acc[r] + cb) * scale + shift;
    float s = y * sigm_(y);
    merged[(size_t)row * DD + HH + ch] = s;
    merged_bf[(size_t)row * DD + HH + ch] = (__bf16)s;
  }
}

// ---------------------------------------------------------------------------
// Sequential LSTM scan, all 4 batches in one persistent workgroup.
// h (bf16) / c (f32) live in LDS.  Per step: g = h @ w_hh^T via WMMA
// (16 waves x 4 N-tiles, K-block-outer so the LDS h-fragment is loaded once
// per kb and feeds 4 accumulators), while the NEXT step's gx slice streams in
// via async global->LDS (double buffered, s_wait_asynccnt fenced).
// ---------------------------------------------------------------------------
__global__ void lstm_scan(const float* __restrict__ gx,     // (B,T,1024), biases pre-added
                          const __bf16* __restrict__ whh,   // (1024,256) bf16
                          float* __restrict__ hout, int hstride,      // optional f32 out
                          __bf16* __restrict__ houtb, int hbstride) { // bf16 out
  __shared__ __bf16 h_bf[16][HH];        // rows 0..3 = batches, 4..15 stay zero
  __shared__ float  c_s[BB][HH];
  __shared__ float  g_s[BB][GATES];
  __shared__ float  gx_s[2][BB * GATES]; // async double buffer (32KB)

  int tid = threadIdx.x;
  for (int i = tid; i < 16 * HH; i += 512) ((__bf16*)h_bf)[i] = (__bf16)0.0f;
  for (int i = tid; i < BB * HH; i += 512) ((float*)c_s)[i] = 0.f;

  // prefetch gx[t=0] into buffer 0: each thread moves 8 floats = 2x b128
  {
    int u0 = tid * 8;
    int b = u0 >> 10, col = u0 & (GATES - 1);
    const float* g0 = gx + (size_t)(b * TT) * GATES + col;
    unsigned l0 = (unsigned)(uintptr_t)&gx_s[0][u0];
    async_copy_b128(l0, g0);
    async_copy_b128(l0 + 16, g0 + 4);
  }
  asm volatile("s_wait_asynccnt 0" ::: "memory");
  __syncthreads();

  int wave = tid >> 5, lane = tid & 31, lo = lane & 15, hi = lane >> 4;
  const __bf16* brow[4];
#pragma unroll
  for (int u = 0; u < 4; ++u)
    brow[u] = whh + (size_t)((wave * 4 + u) * 16 + lo) * HH;

  for (int t = 0; t < TT; ++t) {
    // --- recurrent GEMM: g[4,1024] = h[4,256] @ whh^T (M padded to 16) ---
    v8f acc[4] = {};
#pragma unroll
    for (int kb = 0; kb < 8; ++kb) {
      int k0 = kb * 32 + hi * 8;
      v16bf a = frag_bf16(&h_bf[lo][0], k0);     // one LDS fetch feeds 4 wmmas
      v16bf b0 = frag_bf16(brow[0], k0);
      v16bf b1 = frag_bf16(brow[1], k0);
      v16bf b2 = frag_bf16(brow[2], k0);
      v16bf b3 = frag_bf16(brow[3], k0);
      acc[0] = __builtin_amdgcn_wmma_f32_16x16x32_bf16(false, a, false, b0, (short)0, acc[0], false, false);
      acc[1] = __builtin_amdgcn_wmma_f32_16x16x32_bf16(false, a, false, b1, (short)0, acc[1], false, false);
      acc[2] = __builtin_amdgcn_wmma_f32_16x16x32_bf16(false, a, false, b2, (short)0, acc[2], false, false);
      acc[3] = __builtin_amdgcn_wmma_f32_16x16x32_bf16(false, a, false, b3, (short)0, acc[3], false, false);
    }
    if (hi == 0) {                      // rows 0..3 live in vgprs 0..3, lanes 0..15
#pragma unroll
      for (int u = 0; u < 4; ++u)
#pragma unroll
        for (int r = 0; r < BB; ++r) g_s[r][(wave * 4 + u) * 16 + lo] = acc[u][r];
    }

    // --- overlap: kick off async copy of gx[t+1], then fence step-t's copy ---
    if (t + 1 < TT) {
      int u0 = tid * 8;
      int b = u0 >> 10, col = u0 & (GATES - 1);
      const float* g0 = gx + (size_t)(b * TT + (t + 1)) * GATES + col;
      unsigned l0 = (unsigned)(uintptr_t)&gx_s[(t + 1) & 1][u0];
      async_copy_b128(l0, g0);
      async_copy_b128(l0 + 16, g0 + 4);
      asm volatile("s_wait_asynccnt 2" ::: "memory");  // step-t slice resident
    } else {
      asm volatile("s_wait_asynccnt 0" ::: "memory");
    }
    __syncthreads();

    // --- fused gates: i,f,g,o -> c,h ---
    const float* gxb = &gx_s[t & 1][0];
#pragma unroll
    for (int rep = 0; rep < 2; ++rep) {
      int u = tid + rep * 512;
      int b = u >> 8, j = u & 255;
      const float* gxr = gxb + b * GATES;
      float gi = g_s[b][j]          + gxr[j];
      float gf = g_s[b][HH + j]     + gxr[HH + j];
      float gg = g_s[b][2 * HH + j] + gxr[2 * HH + j];
      float go = g_s[b][3 * HH + j] + gxr[3 * HH + j];
      float c = sigm_(gf) * c_s[b][j] + sigm_(gi) * tanhf(gg);
      float h = sigm_(go) * tanhf(c);
      c_s[b][j] = c;
      h_bf[b][j] = (__bf16)h;
      houtb[(size_t)(b * TT + t) * hbstride + j] = (__bf16)h;
      if (hout) hout[(size_t)(b * TT + t) * hstride + j] = h;
    }
    __syncthreads();
  }
}

// ---------------------------------------------------------------------------
// Attention, last query row only. One block per (b, head).
// ---------------------------------------------------------------------------
__global__ void attention_last(const float* __restrict__ merged,
                               const float* __restrict__ kv,     // (B,T,[k512|v512])
                               const float* __restrict__ qkv_w,
                               const float* __restrict__ qkv_b,
                               float* __restrict__ attn_last) {
  __shared__ float q_s[64];
  __shared__ float s_s[TT];
  __shared__ float red[256];
  __shared__ float pv[4 * 64];
  int b = blockIdx.x >> 3, h = blockIdx.x & 7;
  int tid = threadIdx.x;

  if (tid < 64) {                                        // q at t = T-1, pre-scaled
    const float* mrow = merged + (size_t)(b * TT + TT - 1) * DD;
    const float* wrow = qkv_w + (size_t)(h * 64 + tid) * DD;
    float acc = qkv_b[h * 64 + tid];
    for (int k = 0; k < DD; ++k) acc += mrow[k] * wrow[k];
    q_s[tid] = acc * 0.125f;                             // hd^-0.5, hd=64
  }
  __syncthreads();

  float lmax = -3.4e38f;
  for (int t = tid; t < TT; t += 256) {
    const float* krow = kv + (size_t)(b * TT + t) * GATES + h * 64;
    __builtin_prefetch(krow + 256 * GATES, 0, 3);
    float s = 0.f;
#pragma unroll 16
    for (int d = 0; d < 64; ++d) s += q_s[d] * krow[d];
    s_s[t] = s;
    lmax = fmaxf(lmax, s);
  }
  red[tid] = lmax; __syncthreads();
  for (int st = 128; st > 0; st >>= 1) { if (tid < st) red[tid] = fmaxf(red[tid], red[tid + st]); __syncthreads(); }
  float mx = red[0]; __syncthreads();

  float lsum = 0.f;
  for (int t = tid; t < TT; t += 256) { float p = expf(s_s[t] - mx); s_s[t] = p; lsum += p; }
  red[tid] = lsum; __syncthreads();
  for (int st = 128; st > 0; st >>= 1) { if (tid < st) red[tid] += red[tid + st]; __syncthreads(); }
  float inv = 1.f / red[0]; __syncthreads();

  int d = tid & 63, ck = tid >> 6;
  float acc = 0.f;
  for (int t = ck * 512; t < ck * 512 + 512; ++t)
    acc += s_s[t] * kv[(size_t)(b * TT + t) * GATES + DD + h * 64 + d];
  pv[ck * 64 + d] = acc; __syncthreads();
  if (tid < 64)
    attn_last[b * DD + h * 64 + tid] = (pv[tid] + pv[64 + tid] + pv[128 + tid] + pv[192 + tid]) * inv;
}

// ---------------------------------------------------------------------------
// proj + residual + LayerNorm + fc1/SiLU + fc2 + heads. One block, loop b.
// ---------------------------------------------------------------------------
__global__ void head_kernel(const float* __restrict__ attn_last,
                            const float* __restrict__ merged,
                            const float* __restrict__ proj_w, const float* __restrict__ proj_b,
                            const float* __restrict__ ln_g,  const float* __restrict__ ln_b,
                            const float* __restrict__ fc1_w, const float* __restrict__ fc1_b,
                            const float* __restrict__ fc2_w, const float* __restrict__ fc2_b,
                            float* __restrict__ out) {
  __shared__ float z_s[DD];
  __shared__ float f_s[HH];
  __shared__ float red[DD];
  int tid = threadIdx.x;
  for (int b = 0; b < BB; ++b) {
    const float* al = attn_last + b * DD;
    const float* ml = merged + (size_t)(b * TT + TT - 1) * DD;
    float ctx = proj_b[tid] + ml[tid];
    const float* wr = proj_w + (size_t)tid * DD;
    for (int k = 0; k < DD; ++k) ctx += al[k] * wr[k];

    red[tid] = ctx; __syncthreads();
    for (int s = 256; s > 0; s >>= 1) { if (tid < s) red[tid] += red[tid + s]; __syncthreads(); }
    float mu = red[0] * (1.0f / DD); __syncthreads();
    float dz = ctx - mu;
    red[tid] = dz * dz; __syncthreads();
    for (int s = 256; s > 0; s >>= 1) { if (tid < s) red[tid] += red[tid + s]; __syncthreads(); }
    float var = red[0] * (1.0f / DD); __syncthreads();
    z_s[tid] = dz * rsqrtf(var + 1e-5f) * ln_g[tid] + ln_b[tid];
    __syncthreads();

    if (tid < HH) {
      float a = fc1_b[tid];
      const float* w = fc1_w + (size_t)tid * DD;
      for (int k = 0; k < DD; ++k) a += z_s[k] * w[k];
      f_s[tid] = a * sigm_(a);
    }
    __syncthreads();

    if (tid < 3) {
      float l = fc2_b[tid];
      const float* w = fc2_w + (size_t)tid * HH;
      for (int k = 0; k < HH; ++k) l += f_s[k] * w[k];
      if (tid == 0)      out[b]          = tanhf(l);            // trend
      else if (tid == 1) out[BB + b]     = log1pf(expf(l));     // softplus
      else               out[2 * BB + b] = sigm_(l);            // direction
    }
    __syncthreads();
  }
}

extern "C" void kernel_launch(void* const* d_in, const int* in_sizes, int n_in,
                              void* d_out, int out_size, void* d_ws, size_t ws_size,
                              hipStream_t stream) {
  (void)in_sizes; (void)n_in; (void)out_size; (void)ws_size;
  const float* x       = (const float*)d_in[0];
  const float* w_ih0   = (const float*)d_in[1];
  const float* w_hh0   = (const float*)d_in[2];
  const float* b_ih0   = (const float*)d_in[3];
  const float* b_hh0   = (const float*)d_in[4];
  const float* w_ih1   = (const float*)d_in[5];
  const float* w_hh1   = (const float*)d_in[6];
  const float* b_ih1   = (const float*)d_in[7];
  const float* b_hh1   = (const float*)d_in[8];
  const float* conv_w  = (const float*)d_in[9];
  const float* conv_b  = (const float*)d_in[10];
  const float* bn_g    = (const float*)d_in[11];
  const float* bn_b    = (const float*)d_in[12];
  const float* bn_mean = (const float*)d_in[13];
  const float* bn_var  = (const float*)d_in[14];
  const float* qkv_w   = (const float*)d_in[15];
  const float* qkv_b   = (const float*)d_in[16];
  const float* proj_w  = (const float*)d_in[17];
  const float* proj_b  = (const float*)d_in[18];
  const float* ln_g    = (const float*)d_in[19];
  const float* ln_b    = (const float*)d_in[20];
  const float* fc1_w   = (const float*)d_in[21];
  const float* fc1_b   = (const float*)d_in[22];
  const float* fc2_w   = (const float*)d_in[23];
  const float* fc2_b   = (const float*)d_in[24];
  float* out = (float*)d_out;

  char* ws = (char*)d_ws;
  float*  gx       = (float*)(ws);                  // 32MB  (reused as kv)
  float*  merged   = (float*)(ws + 33554432);       // 16MB
  __bf16* mergedb  = (__bf16*)(ws + 50331648);      //  8MB
  __bf16* h1b      = (__bf16*)(ws + 58720256);      //  4MB
  __bf16* xb       = (__bf16*)(ws + 62914560);      //  1MB
  __bf16* whh0b    = (__bf16*)(ws + 63963136);      // 512KB
  __bf16* whh1b    = (__bf16*)(ws + 64487424);      // 512KB
  __bf16* wih0b    = (__bf16*)(ws + 65011712);      // 128KB
  __bf16* wih1b    = (__bf16*)(ws + 65142784);      // 512KB
  __bf16* kvwb     = (__bf16*)(ws + 65667072);      //  1MB
  __bf16* convwb   = (__bf16*)(ws + 66715648);      // 160KB
  float*  attn_l   = (float*)(ws + 66879488);       //  8KB
  float*  kv       = gx;                            // gx dead after the scans

  // one-time bf16 packs (bandwidth-trivial)
  pack_bf16<<<2048, 256, 0, stream>>>(x, xb, BB * TT * INF_);
  pack_bf16<<<1024, 256, 0, stream>>>(w_hh0, whh0b, GATES * HH);
  pack_bf16<<<1024, 256, 0, stream>>>(w_hh1, whh1b, GATES * HH);
  pack_bf16<<<256,  256, 0, stream>>>(w_ih0, wih0b, GATES * INF_);
  pack_bf16<<<1024, 256, 0, stream>>>(w_ih1, wih1b, GATES * HH);
  pack_bf16<<<2048, 256, 0, stream>>>(qkv_w + 512 * DD, kvwb, GATES * DD);
  pack_convw<<<320, 256, 0, stream>>>(conv_w, convwb);

  // gx0 = x @ w_ih0^T + b_ih0 + b_hh0   (Mtiles=512, Ntiles=64 -> 8192 waves)
  gemm_bf16_wmma<<<1024, 256, 0, stream>>>(xb, INF_, wih0b, INF_, b_ih0, b_hh0,
                                           gx, GATES, 512, 64, INF_);
  lstm_scan<<<1, 512, 0, stream>>>(gx, whh0b, nullptr, 0, h1b, HH);

  // gx1 = h1 @ w_ih1^T + b_ih1 + b_hh1  (K=256)
  gemm_bf16_wmma<<<1024, 256, 0, stream>>>(h1b, HH, wih1b, HH, b_ih1, b_hh1,
                                           gx, GATES, 512, 64, HH);
  lstm_scan<<<1, 512, 0, stream>>>(gx, whh1b, merged, DD, mergedb, DD);  // merged[:, :256]

  conv_bn_silu_wmma<<<1024, 256, 0, stream>>>(xb, convwb, conv_b, bn_g, bn_b,
                                              bn_mean, bn_var, merged, mergedb);  // [:, 256:]

  // kv = merged @ qkv_w[512:1536]^T + qkv_b[512:]  (rows 512..1023=k, 1024..1535=v)
  gemm_bf16_wmma<<<1024, 256, 0, stream>>>(mergedb, DD, kvwb, DD,
                                           qkv_b + 512, nullptr, kv, GATES, 512, 64, DD);

  attention_last<<<32, 256, 0, stream>>>(merged, kv, qkv_w, qkv_b, attn_l);
  head_kernel<<<1, 512, 0, stream>>>(attn_l, merged, proj_w, proj_b, ln_g, ln_b,
                                     fc1_w, fc1_b, fc2_w, fc2_b, out);
}
